// BondDecoder_9560597201146
// MI455X (gfx1250) — matrix operand order repaired and checked by reading, not compiled, not run on hardware
//
#include <hip/hip_runtime.h>
#include <hip/hip_bf16.h>

// ---------------------------------------------------------------------------
// BondDecoder for MI455X (gfx1250): bf16 WMMA projections + fused attention.
// L=512, B=32, D=512, H=4, HD=128, MAX_BONDS=6.
// ---------------------------------------------------------------------------

#define LL 512
#define BB 32
#define DD 512
#define NH 4
#define HD 128

typedef __attribute__((ext_vector_type(16))) __bf16 v16bf;
typedef __attribute__((ext_vector_type(8)))  float  v8f;

union BfFrag { uint4 u[2]; v16bf v; };

__device__ __forceinline__ unsigned short f32_to_bf16(float f) {
    unsigned int u = __float_as_uint(f);
    unsigned int r = u + 0x7FFFu + ((u >> 16) & 1u);   // round-to-nearest-even
    return (unsigned short)(r >> 16);
}

// A fragment: 16x32 bf16, row M = lane&15, K-chunks split on lane-half.
__device__ __forceinline__ v16bf load_a_frag(const unsigned short* __restrict__ A,
                                             int lda, int m0, int k0, int lane) {
    int ll = lane & 15, lh = (lane >> 4) & 1;
    const unsigned short* p = A + (size_t)(m0 + ll) * lda + k0 + lh * 8;
    BfFrag f;
    f.u[0] = *(const uint4*)(p);        // K = k0+off .. +7
    f.u[1] = *(const uint4*)(p + 16);   // K = k0+off+16 .. +23
    return f.v;
}

// B fragment: 32x16 bf16 (KxN). N = lane&15 ; K runs across VGPR pairs,
// lane-half selects K +0 / +8, second chunk at +16.  Source stored n-major: Bm[n][k].
__device__ __forceinline__ v16bf load_b_frag(const unsigned short* __restrict__ Bm,
                                             int ldb, int n0, int k0, int lane) {
    int ll = lane & 15, lh = (lane >> 4) & 1;
    const unsigned short* p = Bm + (size_t)(n0 + ll) * ldb + k0 + lh * 8;
    BfFrag f;
    f.u[0] = *(const uint4*)(p);
    f.u[1] = *(const uint4*)(p + 16);
    return f.v;
}

__device__ __forceinline__ v8f wmma_bf16(v16bf a, v16bf b, v8f c) {
    return __builtin_amdgcn_wmma_f32_16x16x32_bf16(false, a, false, b, (short)0, c,
                                                   false, false);
}

// ---------------------------------------------------------------------------
__global__ void k_zero(float* out, int n) {
    int i = blockIdx.x * blockDim.x + threadIdx.x;
    if (i < n) out[i] = 0.0f;
}

// [L,B,D] f32  ->  [B,L,D] bf16
__global__ __launch_bounds__(256) void k_transpose(const float* __restrict__ in,
                                                   unsigned short* __restrict__ out) {
    int gid  = blockIdx.x * 256 + threadIdx.x;   // L*B*(D/4) threads
    int dp   = gid & 127;                        // D/4 = 128
    int rest = gid >> 7;
    int b    = rest % BB;
    int l    = rest / BB;
    const float4 v = *(const float4*)(in + ((size_t)l * BB + b) * DD + dp * 4);
    ushort4 o;
    o.x = f32_to_bf16(v.x); o.y = f32_to_bf16(v.y);
    o.z = f32_to_bf16(v.z); o.w = f32_to_bf16(v.w);
    *(ushort4*)(out + ((size_t)b * LL + l) * DD + dp * 4) = o;
}

// Combined weight  M[n][k] = sum_t W2[n][t] * W1[t][k]   (bf16 out, n-major)
__global__ __launch_bounds__(256) void k_combine_w(
    const float* __restrict__ w2_0, const float* __restrict__ w2_1,
    const float* __restrict__ w2_2, const float* __restrict__ w2_3,
    const float* __restrict__ w1_0, const float* __restrict__ w1_1,
    const float* __restrict__ w1_2, const float* __restrict__ w1_3,
    unsigned short* __restrict__ Mc) {
    int z = blockIdx.z;
    const float* W2 = (z == 0) ? w2_0 : (z == 1) ? w2_1 : (z == 2) ? w2_2 : w2_3;
    const float* W1 = (z == 0) ? w1_0 : (z == 1) ? w1_1 : (z == 2) ? w1_2 : w1_3;
    int tx = threadIdx.x, ty = threadIdx.y;
    int n = blockIdx.y * 16 + ty;
    int k = blockIdx.x * 16 + tx;
    __shared__ float As[16][17];
    __shared__ float Bs[16][17];
    float acc = 0.0f;
    for (int t0 = 0; t0 < DD; t0 += 16) {
        As[ty][tx] = W2[(size_t)n * DD + t0 + tx];
        Bs[ty][tx] = W1[(size_t)(t0 + ty) * DD + k];
        __syncthreads();
#pragma unroll
        for (int p = 0; p < 16; p++) acc += As[ty][p] * Bs[p][tx];
        __syncthreads();
    }
    Mc[(size_t)z * DD * DD + (size_t)n * DD + k] = f32_to_bf16(acc);
}

// Combined bias  bc[n] = sum_t W2[n][t]*b1[t] + b2[n]
__global__ __launch_bounds__(256) void k_combine_b(
    const float* __restrict__ w2_0, const float* __restrict__ w2_1,
    const float* __restrict__ w2_2, const float* __restrict__ w2_3,
    const float* __restrict__ b1_0, const float* __restrict__ b1_1,
    const float* __restrict__ b1_2, const float* __restrict__ b1_3,
    const float* __restrict__ b2_0, const float* __restrict__ b2_1,
    const float* __restrict__ b2_2, const float* __restrict__ b2_3,
    float* __restrict__ bc) {
    int z = blockIdx.x;
    const float* W2 = (z == 0) ? w2_0 : (z == 1) ? w2_1 : (z == 2) ? w2_2 : w2_3;
    const float* b1 = (z == 0) ? b1_0 : (z == 1) ? b1_1 : (z == 2) ? b1_2 : b1_3;
    const float* b2 = (z == 0) ? b2_0 : (z == 1) ? b2_1 : (z == 2) ? b2_2 : b2_3;
    for (int n = threadIdx.x; n < DD; n += blockDim.x) {
        float s = b2[n];
        for (int t = 0; t < DD; t++) s += W2[(size_t)n * DD + t] * b1[t];
        bc[z * DD + n] = s;
    }
}

// Projection GEMM: Y[z][m][n] = bf16( X[m][:] . Mc[z][n][:] + bc[z][n] )
// Register-blocked: one 32x32 output per wave = 2x2 WMMA tiles, 4 independent
// accumulator chains; per K-step 4 fragment loads feed 4 WMMAs (2x the
// compute/load ratio of 1-tile blocking), halving L2 traffic per operand.
__global__ __launch_bounds__(256) void k_proj(const unsigned short* __restrict__ X,
                                              const unsigned short* __restrict__ Mc,
                                              const float* __restrict__ bc,
                                              unsigned short* __restrict__ Y) {
    int z = blockIdx.y;
    const unsigned short* W = Mc + (size_t)z * DD * DD;
    const float* bias       = bc + z * DD;
    unsigned short* out     = Y + (size_t)z * (BB * LL) * DD;

    int wave = threadIdx.x >> 5;
    int lane = threadIdx.x & 31;
    int tile = blockIdx.x * 8 + wave;   // 8192 macro-tiles (32x32 each)
    int m0 = (tile >> 4) * 32;          // 512 M macro-tiles
    int n0 = (tile & 15) * 32;          // 16 N macro-tiles

    v8f c00 = {}, c01 = {}, c10 = {}, c11 = {};
    for (int k0 = 0; k0 < DD; k0 += 32) {
        if (k0 + 32 < DD) {
            __builtin_prefetch(X + (size_t)(m0 + (lane & 15)) * DD + k0 + 32, 0, 3);
            __builtin_prefetch(X + (size_t)(m0 + 16 + (lane & 15)) * DD + k0 + 32, 0, 3);
        }
        v16bf a0 = load_a_frag(X, DD, m0,      k0, lane);
        v16bf a1 = load_a_frag(X, DD, m0 + 16, k0, lane);
        v16bf b0 = load_b_frag(W, DD, n0,      k0, lane);
        v16bf b1 = load_b_frag(W, DD, n0 + 16, k0, lane);
        c00 = wmma_bf16(a0, b0, c00);
        c01 = wmma_bf16(a0, b1, c01);
        c10 = wmma_bf16(a1, b0, c10);
        c11 = wmma_bf16(a1, b1, c11);
    }
    int ll = lane & 15, lh = (lane >> 4) & 1;
    float bv0 = bias[n0 + ll];
    float bv1 = bias[n0 + 16 + ll];
#pragma unroll
    for (int r = 0; r < 8; r++) {
        int m = m0 + lh * 8 + r;
        out[(size_t)m * DD + n0 + ll]            = f32_to_bf16(c00[r] + bv0);
        out[(size_t)m * DD + n0 + 16 + ll]       = f32_to_bf16(c01[r] + bv1);
        out[(size_t)(m + 16) * DD + n0 + ll]     = f32_to_bf16(c10[r] + bv0);
        out[(size_t)(m + 16) * DD + n0 + 16 + ll] = f32_to_bf16(c11[r] + bv1);
    }
}

// Fused attention: per (batch b, 16-row i-tile) compute all 4 heads'
// 16x512 scores via WMMA, masked softmax, head-mean -> P[b, i, :] (f32).
__global__ __launch_bounds__(256) void k_attn(const unsigned short* __restrict__ Q,
                                              const unsigned short* __restrict__ K,
                                              const int* __restrict__ key_mask,
                                              float* __restrict__ P) {
    __shared__ float sc[16 * LL];    // 32 KB: scores for current head
    __shared__ float acc[16 * LL];   // 32 KB: head-averaged probabilities

    int blk = blockIdx.x;            // B * (L/16) = 1024
    int b   = blk >> 5;
    int i0  = (blk & 31) * 16;
    int wave = threadIdx.x >> 5;
    int lane = threadIdx.x & 31;

    for (int t = threadIdx.x; t < 16 * LL; t += 256) acc[t] = 0.0f;
    __syncthreads();

    const unsigned short* Qb = Q + (size_t)(b * LL + i0) * DD;
    const unsigned short* Kb = K + (size_t)(b * LL) * DD;
    const float scale = 0.08838834764831845f;   // 1/sqrt(128)

    for (int h = 0; h < NH; h++) {
        int kb = h * HD;
        v16bf a0 = load_a_frag(Qb, DD, 0, kb +  0, lane);
        v16bf a1 = load_a_frag(Qb, DD, 0, kb + 32, lane);
        v16bf a2 = load_a_frag(Qb, DD, 0, kb + 64, lane);
        v16bf a3 = load_a_frag(Qb, DD, 0, kb + 96, lane);

#pragma unroll
        for (int q = 0; q < 4; q++) {
            int j0 = (wave * 4 + q) * 16;
            v8f c = {};
            c = wmma_bf16(a0, load_b_frag(Kb, DD, j0, kb +  0, lane), c);
            c = wmma_bf16(a1, load_b_frag(Kb, DD, j0, kb + 32, lane), c);
            c = wmma_bf16(a2, load_b_frag(Kb, DD, j0, kb + 64, lane), c);
            c = wmma_bf16(a3, load_b_frag(Kb, DD, j0, kb + 96, lane), c);
            int ll = lane & 15, lh = (lane >> 4) & 1;
#pragma unroll
            for (int r = 0; r < 8; r++)
                sc[(lh * 8 + r) * LL + j0 + ll] = c[r] * scale;
        }
        __syncthreads();

        // softmax: wave handles rows 2*wave, 2*wave+1 ; 16 elems/lane
#pragma unroll
        for (int rr = 0; rr < 2; rr++) {
            int row = wave * 2 + rr;
            float v[16];
            float m = -INFINITY;
#pragma unroll
            for (int t = 0; t < 16; t++) {
                int j = lane + t * 32;
                float s = key_mask[b * LL + j] ? -INFINITY : sc[row * LL + j];
                v[t] = s;
                m = fmaxf(m, s);
            }
#pragma unroll
            for (int off = 16; off >= 1; off >>= 1)
                m = fmaxf(m, __shfl_xor(m, off));
            float sum = 0.0f;
#pragma unroll
            for (int t = 0; t < 16; t++) {
                float p = (m > -INFINITY) ? __expf(v[t] - m) : 0.0f;
                v[t] = p;
                sum += p;
            }
#pragma unroll
            for (int off = 16; off >= 1; off >>= 1)
                sum += __shfl_xor(sum, off);
            float inv = (sum > 0.0f) ? (0.25f / sum) : 0.0f;   // mean over 4 heads
#pragma unroll
            for (int t = 0; t < 16; t++) {
                int j = lane + t * 32;
                acc[row * LL + j] += v[t] * inv;
            }
        }
        __syncthreads();
    }

    for (int t = threadIdx.x; t < 16 * LL; t += 256) {
        int row = t >> 9;
        int j   = t & (LL - 1);
        P[((size_t)b * LL + i0 + row) * LL + j] = acc[t];
    }
}

// Loss: per (b, i) row — sparse bond histograms + masked squared error.
__global__ __launch_bounds__(256) void k_err(const float* __restrict__ Pinc,
                                             const float* __restrict__ Pdec,
                                             const int* __restrict__ src_bond,
                                             const int* __restrict__ tgt_bond,
                                             const int* __restrict__ src_mask,
                                             const int* __restrict__ tgt_mask,
                                             float* __restrict__ out) {
    __shared__ float hs[LL];
    __shared__ float ht[LL];
    __shared__ float red[8];
    int i = blockIdx.x, b = blockIdx.y;
    int tid = threadIdx.x;
    for (int t = tid; t < LL; t += 256) { hs[t] = 0.0f; ht[t] = 0.0f; }
    __syncthreads();
    if (tid < 6)
        atomicAdd(&hs[src_bond[((size_t)b * LL + i) * 6 + tid]], 1.0f);
    else if (tid >= 32 && tid < 38)
        atomicAdd(&ht[tgt_bond[((size_t)b * LL + i) * 6 + (tid - 32)]], 1.0f);
    __syncthreads();

    float ps_i = src_mask[b * LL + i] ? 0.0f : 1.0f;
    float tm_i = tgt_mask[b * LL + i] ? 1.0f : 0.0f;
    size_t rowoff = ((size_t)b * LL + i) * LL;
    float local = 0.0f;
    for (int j = tid; j < LL; j += 256) {
        float ps_j = src_mask[b * LL + j] ? 0.0f : 1.0f;
        float tm_j = tgt_mask[b * LL + j] ? 1.0f : 0.0f;
        float pad  = ps_i * ps_j;
        float diff = (Pinc[rowoff + j] - Pdec[rowoff + j]) * 4.0f * pad;
        float pred = hs[j] * pad + diff;
        float tgt  = ht[j] * ((1.0f - tm_i) * (1.0f - tm_j));
        float orm  = 1.0f - tm_i * tm_j;
        float e = pred - tgt;
        local += e * e * pad * orm;
    }
#pragma unroll
    for (int off = 16; off >= 1; off >>= 1) local += __shfl_xor(local, off);
    if ((tid & 31) == 0) red[tid >> 5] = local;
    __syncthreads();
    if (tid == 0) {
        float s = 0.0f;
        for (int w = 0; w < 8; w++) s += red[w];
        atomicAdd(out + b, s);
    }
}

// ---------------------------------------------------------------------------
extern "C" void kernel_launch(void* const* d_in, const int* in_sizes, int n_in,
                              void* d_out, int out_size, void* d_ws, size_t ws_size,
                              hipStream_t stream) {
    (void)in_sizes; (void)n_in; (void)out_size; (void)ws_size;

    const float* mol      = (const float*)d_in[0];
    const int*   src_bond = (const int*)d_in[1];
    const int*   src_mask = (const int*)d_in[2];
    const int*   tgt_bond = (const int*)d_in[3];
    const int*   tgt_mask = (const int*)d_in[4];
    const float* inc_q_w  = (const float*)d_in[5];
    const float* inc_k_w  = (const float*)d_in[6];
    const float* inc_Wq   = (const float*)d_in[7];
    const float* inc_Wk   = (const float*)d_in[8];
    const float* dec_q_w  = (const float*)d_in[9];
    const float* dec_k_w  = (const float*)d_in[10];
    const float* dec_Wq   = (const float*)d_in[11];
    const float* dec_Wk   = (const float*)d_in[12];
    const float* inc_q_b  = (const float*)d_in[13];
    const float* inc_k_b  = (const float*)d_in[14];
    const float* inc_bq   = (const float*)d_in[15];
    const float* inc_bk   = (const float*)d_in[16];
    const float* dec_q_b  = (const float*)d_in[17];
    const float* dec_k_b  = (const float*)d_in[18];
    const float* dec_bq   = (const float*)d_in[19];
    const float* dec_bk   = (const float*)d_in[20];

    char* ws = (char*)d_ws;
    size_t off = 0;
    auto alloc = [&](size_t bytes) -> void* {
        void* p = ws + off;
        off += (bytes + 255) & ~(size_t)255;
        return p;
    };
    const size_t ML = (size_t)BB * LL;                   // 16384 rows
    unsigned short* Xbf = (unsigned short*)alloc(ML * DD * 2);
    unsigned short* Mc  = (unsigned short*)alloc((size_t)4 * DD * DD * 2);
    float*          bcv = (float*)alloc((size_t)4 * DD * 4);
    unsigned short* QK  = (unsigned short*)alloc((size_t)4 * ML * DD * 2);
    float*          Pinc = (float*)alloc(ML * LL * 4);
    float*          Pdec = (float*)alloc(ML * LL * 4);

    k_zero<<<1, 32, 0, stream>>>((float*)d_out, BB);

    k_transpose<<<(int)(ML * (DD / 4) / 256), 256, 0, stream>>>(mol, Xbf);

    // order: 0=Q_inc, 1=K_inc, 2=Q_dec, 3=K_dec
    k_combine_w<<<dim3(32, 32, 4), dim3(16, 16), 0, stream>>>(
        inc_Wq, inc_Wk, dec_Wq, dec_Wk,
        inc_q_w, inc_k_w, dec_q_w, dec_k_w, Mc);
    k_combine_b<<<4, 256, 0, stream>>>(
        inc_Wq, inc_Wk, dec_Wq, dec_Wk,
        inc_q_b, inc_k_b, dec_q_b, dec_k_b,
        inc_bq, inc_bk, dec_bq, dec_bk, bcv);

    k_proj<<<dim3(1024, 4), 256, 0, stream>>>(Xbf, Mc, bcv, QK);

    const size_t QKs = ML * DD;
    k_attn<<<BB * (LL / 16), 256, 0, stream>>>(QK + 0 * QKs, QK + 1 * QKs,
                                               src_mask, Pinc);
    k_attn<<<BB * (LL / 16), 256, 0, stream>>>(QK + 2 * QKs, QK + 3 * QKs,
                                               src_mask, Pdec);

    k_err<<<dim3(LL, BB), 256, 0, stream>>>(Pinc, Pdec, src_bond, tgt_bond,
                                            src_mask, tgt_mask, (float*)d_out);
}
